// RETAIN_59124519797162
// MI455X (gfx1250) — matrix operand order repaired and checked by reading, not compiled, not run on hardware
//
#include <hip/hip_runtime.h>
#include <hip/hip_bf16.h>
#include <stdint.h>

// ---------------------------------------------------------------------------
// Problem constants (from the reference)
// ---------------------------------------------------------------------------
constexpr int Bq   = 32;
constexpr int Lq   = 64;
constexpr int Vq   = 10000;
constexpr int Eq   = 128;
constexpr int Hq   = 128;
constexpr int Kq   = 63;          // L - 1
constexpr int NBPq = Bq * Kq;     // 2016 (b,p) pairs
constexpr int H3q  = 3 * Hq;      // 384

// ---------------------------------------------------------------------------
// Vector types
// ---------------------------------------------------------------------------
typedef __attribute__((ext_vector_type(16))) __bf16 bf16x16;
typedef __attribute__((ext_vector_type(8)))  float  floatx8;
typedef __attribute__((ext_vector_type(4)))  int    i32x4;
typedef __attribute__((ext_vector_type(8)))  int    i32x8;
typedef __attribute__((ext_vector_type(4)))  float  f32x4;

// ---------------------------------------------------------------------------
// bf16 helpers
// ---------------------------------------------------------------------------
__device__ inline unsigned int bfbits(float f) {   // RNE f32 -> bf16 bits
  unsigned int u = __builtin_bit_cast(unsigned int, f);
  return (u + 0x7FFFu + ((u >> 16) & 1u)) >> 16;
}
__device__ inline __bf16 f2bf(float f) {
  return __builtin_bit_cast(__bf16, (unsigned short)bfbits(f));
}
// pack 8 fp32 -> 8 bf16 (as i32x4, element 0 in low half of word 0)
__device__ inline i32x4 pack8_bf(f32x4 a, f32x4 b) {
  i32x4 o;
  o[0] = (int)(bfbits(a[0]) | (bfbits(a[1]) << 16));
  o[1] = (int)(bfbits(a[2]) | (bfbits(a[3]) << 16));
  o[2] = (int)(bfbits(b[0]) | (bfbits(b[1]) << 16));
  o[3] = (int)(bfbits(b[2]) | (bfbits(b[3]) << 16));
  return o;
}
__device__ inline bf16x16 combine(i32x4 lo, i32x4 hi) {
  i32x8 w;
  w[0] = lo[0]; w[1] = lo[1]; w[2] = lo[2]; w[3] = lo[3];
  w[4] = hi[0]; w[5] = hi[1]; w[6] = hi[2]; w[7] = hi[3];
  return __builtin_bit_cast(bf16x16, w);
}

// ---------------------------------------------------------------------------
// WMMA fragment loaders (CDNA5 wave32, v_wmma_f32_16x16x32_bf16)
// All addresses are 16B-aligned by construction -> b128 loads.
// ---------------------------------------------------------------------------
// A fragment: 16x32 (MxK) bf16, rp = this lane's row at k=0. Lanes 0-15 hold
// K in {0..7}U{16..23}; lanes 16-31 hold K in {8..15}U{24..31}.
__device__ inline bf16x16 a_frag_bf(const __bf16* rp) {
  const int kh = (((threadIdx.x >> 4) & 1) << 3);
  i32x4 lo = *(const i32x4*)(rp + kh);
  i32x4 hi = *(const i32x4*)(rp + 16 + kh);
  return combine(lo, hi);
}
// Row-guarded A fragment (caller clamps rp in-bounds); select whole vectors.
__device__ inline bf16x16 a_frag_bf_guard(const __bf16* rp, bool rowvalid) {
  const int kh = (((threadIdx.x >> 4) & 1) << 3);
  i32x4 lo = *(const i32x4*)(rp + kh);
  i32x4 hi = *(const i32x4*)(rp + 16 + kh);
  i32x4 z = {};
  return combine(rowvalid ? lo : z, rowvalid ? hi : z);
}
// K-tail (exactly 16 valid K): low 8 K always valid, high 8 always invalid.
__device__ inline bf16x16 a_frag_bf_tail16(const __bf16* rp) {
  const int kh = (((threadIdx.x >> 4) & 1) << 3);
  i32x4 lo = *(const i32x4*)(rp + kh);
  i32x4 z = {};
  return combine(lo, z);
}

// B fragment: 32x16 (KxN) with B[k][n] = W[n][k]; W row-major [Ntot, Kdim],
// Wt = W + n0*ldw + kk. lane -> column (lane%16), half-wave -> K {0..7}U{16..23}/{8..15}U{24..31}.
__device__ inline bf16x16 b_frag_wT(const __bf16* Wt, int ldw) {
  const int lane = threadIdx.x & 31;
  const __bf16* cp = Wt + (size_t)(lane & 15) * ldw + (((lane >> 4) & 1) << 3);
  i32x4 lo = *(const i32x4*)cp;
  i32x4 hi = *(const i32x4*)(cp + 16);
  return combine(lo, hi);
}
// K-tail (16 valid K): low 8 K (k=8h+s) always valid, high 8 always zero.
// Address stays within k0..15 so the load never leaves the row.
__device__ inline bf16x16 b_frag_wT_tail16(const __bf16* Wt, int ldw) {
  const int lane = threadIdx.x & 31;
  const __bf16* cp = Wt + (size_t)(lane & 15) * ldw + (((lane >> 4) & 1) << 3);
  i32x4 lo = *(const i32x4*)cp;
  i32x4 z = {};
  return combine(lo, z);
}

__device__ inline floatx8 wmma_bf16(bf16x16 a, bf16x16 b, floatx8 c) {
  return __builtin_amdgcn_wmma_f32_16x16x32_bf16(
      false, a, false, b, (short)0, c, false, false);
}

// Store D fragment: VGPR i -> row r0+i (r0 = 8 for lanes>=16), col = lane%16.
__device__ inline void store_d_f32(float* C, int ldc, floatx8 d) {
  const int lane = threadIdx.x & 31;
  const int col = lane & 15;
  const int r0 = ((lane >> 4) & 1) << 3;
#pragma unroll
  for (int i = 0; i < 8; ++i) C[(size_t)(r0 + i) * ldc + col] = d[i];
}

// ---------------------------------------------------------------------------
// Elementwise kernels
// ---------------------------------------------------------------------------
// fp32 -> bf16, 8 elements/thread, b128 in / b128 out (n must be multiple of 8)
__global__ void convert8_f32_bf16_kernel(const float* __restrict__ src,
                                         __bf16* __restrict__ dst, int n8) {
  int i = blockIdx.x * blockDim.x + threadIdx.x;
  if (i >= n8) return;
  f32x4 a = ((const f32x4*)src)[2 * (size_t)i];
  f32x4 b = ((const f32x4*)src)[2 * (size_t)i + 1];
  ((i32x4*)dst)[i] = pack8_bf(a, b);
}

__global__ void zero_f32_kernel(float* __restrict__ p, int n4) {
  int i = blockIdx.x * blockDim.x + threadIdx.x;
  if (i < n4) { f32x4 z = {}; ((f32x4*)p)[i] = z; }
}

// ---------------------------------------------------------------------------
// Kernel 1: emb = relu(codes @ W_embed^T)    M=2048, N=128, K=10000
// One wave per 16x16 output tile; 8 waves/block; bf16 inputs pre-converted.
// ---------------------------------------------------------------------------
__global__ void embed_gemm_kernel(const __bf16* __restrict__ codes_bf, // [2048,V]
                                  const __bf16* __restrict__ We,       // [E,V]
                                  float* __restrict__ emb,             // [2048,E]
                                  __bf16* __restrict__ emb_bf) {
  const int tid  = threadIdx.x;
  const int wave = tid >> 5;
  const int lane = tid & 31;
  const int tile = blockIdx.x * 8 + wave;     // 0..1023
  const int mt = tile >> 3;                   // 0..127
  const int nt = tile & 7;                    // 0..7

  const __bf16* arp = codes_bf + (size_t)(mt * 16 + (lane & 15)) * Vq;
  const __bf16* Wt0 = We + (size_t)(nt * 16) * Vq;

  floatx8 acc = {};
  int kk = 0;
  for (; kk + 32 <= Vq; kk += 32) {           // 312 iterations
    __builtin_prefetch(arp + kk + 512, 0, 0); // global_prefetch_b8
    bf16x16 a = a_frag_bf(arp + kk);
    bf16x16 b = b_frag_wT(Wt0 + kk, Vq);
    acc = wmma_bf16(a, b, acc);
  }
  {  // compile-time K tail of 16 (10000 = 312*32 + 16): branch-free
    bf16x16 a = a_frag_bf_tail16(arp + kk);
    bf16x16 b = b_frag_wT_tail16(Wt0 + kk, Vq);
    acc = wmma_bf16(a, b, acc);
  }

  const int col = lane & 15;
  const int r0 = ((lane >> 4) & 1) << 3;
#pragma unroll
  for (int i = 0; i < 8; ++i) {
    float v = fmaxf(acc[i], 0.0f);            // relu
    size_t o = (size_t)(mt * 16 + r0 + i) * Eq + nt * 16 + col;
    emb[o] = v;
    emb_bf[o] = f2bf(v);
  }
}

// ---------------------------------------------------------------------------
// Kernel 2: xproj = emb @ W_ih^T + b_ih   for both GRUs (blockIdx.y selects)
// M=2016 (rows (b,i), i<63), N=384, K=128
// ---------------------------------------------------------------------------
__global__ void xproj_gemm_kernel(const __bf16* __restrict__ emb_bf,  // [2048,E]
                                  const __bf16* __restrict__ Wih_a,   // [384,E]
                                  const __bf16* __restrict__ Wih_b,
                                  const float* __restrict__ bih_a,
                                  const float* __restrict__ bih_b,
                                  float* __restrict__ xpa,            // [2016,384]
                                  float* __restrict__ xpb) {
  const int g = blockIdx.y;
  const __bf16* Wih = g ? Wih_b : Wih_a;
  const float*  bih = g ? bih_b : bih_a;
  float* xp = g ? xpb : xpa;

  const int tid  = threadIdx.x;
  const int wave = tid >> 5;
  const int lane = tid & 31;
  const int tile = blockIdx.x * 8 + wave;   // 0..3023 (126 * 24)
  const int mt = tile / 24, nt = tile % 24;

  const int mrow = mt * 16 + (lane & 15);   // 0..2015
  const int bb = mrow / Kq, ii = mrow % Kq;
  const __bf16* arp = emb_bf + (size_t)(bb * Lq + ii) * Eq;

  floatx8 acc = {};
#pragma unroll
  for (int kk = 0; kk < Eq; kk += 32) {
    bf16x16 a = a_frag_bf(arp + kk);
    bf16x16 b = b_frag_wT(Wih + (size_t)(nt * 16) * Eq + kk, Eq);
    acc = wmma_bf16(a, b, acc);
  }

  const int col = lane & 15;
  const int r0 = ((lane >> 4) & 1) << 3;
#pragma unroll
  for (int i = 0; i < 8; ++i) {
    int m = mt * 16 + r0 + i;
    int n = nt * 16 + col;
    xp[(size_t)m * H3q + n] = acc[i] + bih[n];
  }
}

// ---------------------------------------------------------------------------
// Kernel 3: one GRU time step (fused recurrent GEMM + gates).
//   hp = h @ W_hh^T  (32 rows/block, N=384, K=128, bf16 WMMA via LDS)
//   gates -> h_new; GRU a: logit = dot(h_new, W_af)+b_af; GRU b: store h bf16
// Step j: x-input row = m - j (valid when p = m%63 >= j), else x = b_ih.
// ---------------------------------------------------------------------------
__global__ void gru_step_kernel(const float* __restrict__ xproj,   // [2016,384]
                                const __bf16* __restrict__ Whh,    // [384,128] bf16
                                const float* __restrict__ b_ih,    // [384]
                                const float* __restrict__ b_hh,    // [384]
                                float* __restrict__ h,             // [2016,128]
                                const float* __restrict__ W_af,    // [128] (a only)
                                const float* __restrict__ b_af,    // [1]   (a only)
                                float* __restrict__ logits,        // [2016,63] (a)
                                __bf16* __restrict__ hb_all,       // [2016,63,128] (b)
                                int j, int isB) {
  __shared__ __bf16 sA[32 * Hq];        //  8 KB: h tile in bf16
  __shared__ float  sHP[32 * H3q];      // 48 KB: hp tile fp32
  __shared__ float  sRed[32][8];        //  1 KB: logit reduction

  const int tid = threadIdx.x;
  const int m0 = blockIdx.x * 32;

  // stage 32x128 h rows -> bf16 LDS: 16 f32 in (2xb128... 4 loads), 32B out
  {
    const f32x4* hv = (const f32x4*)(h + (size_t)m0 * Hq) + (size_t)tid * 4;
    f32x4 a0 = hv[0], a1 = hv[1], a2 = hv[2], a3 = hv[3];
    i32x4* sv = (i32x4*)sA + (size_t)tid * 2;
    sv[0] = pack8_bf(a0, a1);
    sv[1] = pack8_bf(a2, a3);
  }
  __syncthreads();

  // 48 tiles (2 row-tiles x 24 col-tiles), 6 per wave
  const int wave = tid >> 5;
  const int lane = tid & 31;
  for (int t = wave; t < 48; t += 8) {
    int rt = t / 24, ct = t % 24;
    floatx8 acc = {};
#pragma unroll
    for (int kk = 0; kk < Hq; kk += 32) {
      const __bf16* rp = sA + (size_t)(rt * 16 + (lane & 15)) * Hq + kk;
      bf16x16 a = a_frag_bf(rp);
      bf16x16 b = b_frag_wT(Whh + (size_t)(ct * 16) * Hq + kk, Hq);
      acc = wmma_bf16(a, b, acc);
    }
    store_d_f32(sHP + (size_t)(rt * 16) * H3q + ct * 16, H3q, acc);
  }
  __syncthreads();

  // gate phase: thread -> (row = tid/8, 16 consecutive h-channels)
  const int row = tid >> 3;
  const int sub = tid & 7;
  const int c0 = sub * 16;
  const int m = m0 + row;
  const int p = m % Kq;
  const bool valid = (p >= j);
  const float* xt = valid ? (xproj + (size_t)(m - j) * H3q) : b_ih;
  // (invalid steps feed x=0 through W_ih, i.e. exactly b_ih)

  float hn[16];
  float partial = 0.0f;
#pragma unroll
  for (int i = 0; i < 16; ++i) {
    int c = c0 + i;
    float hpr = sHP[(size_t)row * H3q + c]          + b_hh[c];
    float hpz = sHP[(size_t)row * H3q + Hq + c]     + b_hh[Hq + c];
    float hpn = sHP[(size_t)row * H3q + 2 * Hq + c] + b_hh[2 * Hq + c];
    float xr = xt[c], xz = xt[Hq + c], xn = xt[2 * Hq + c];
    float r = 1.0f / (1.0f + expf(-(xr + hpr)));
    float z = 1.0f / (1.0f + expf(-(xz + hpz)));
    float n = tanhf(xn + r * hpn);
    float hold = h[(size_t)m * Hq + c];
    hn[i] = (1.0f - z) * n + z * hold;
    h[(size_t)m * Hq + c] = hn[i];
    if (!isB) partial += hn[i] * W_af[c];
  }

  if (isB) {   // store bf16 step output as 2x b128
    f32x4 a0 = { hn[0], hn[1], hn[2],  hn[3]  };
    f32x4 a1 = { hn[4], hn[5], hn[6],  hn[7]  };
    f32x4 a2 = { hn[8], hn[9], hn[10], hn[11] };
    f32x4 a3 = { hn[12], hn[13], hn[14], hn[15] };
    i32x4* dv = (i32x4*)(hb_all + ((size_t)m * Kq + j) * Hq + c0);
    dv[0] = pack8_bf(a0, a1);
    dv[1] = pack8_bf(a2, a3);
  } else {     // attention logit: dot(h_new, W_af) + b_af
    sRed[row][sub] = partial;
    __syncthreads();
    if (sub == 0) {
      float s = 0.0f;
#pragma unroll
      for (int q = 0; q < 8; ++q) s += sRed[row][q];
      logits[(size_t)m * Kq + j] = s + b_af[0];
    }
  }
}

// ---------------------------------------------------------------------------
// Kernel 4: per (b,p): softmax over prefix + beta = tanh(h_b @ W_bf^T + b_bf)
// (64x128x128 WMMA GEMM) + context = sum_j alpha*beta*emb[b,p-j]
// ---------------------------------------------------------------------------
__global__ void attn_context_kernel(const float* __restrict__ logits,  // [2016,63]
                                    const __bf16* __restrict__ hb_all, // [2016,63,128]
                                    const __bf16* __restrict__ Wbf,    // [E,H] bf16
                                    const float* __restrict__ b_bf,    // [E]
                                    const float* __restrict__ emb,     // [2048,E]
                                    float* __restrict__ ctx,           // [2016,E]
                                    __bf16* __restrict__ ctx_bf) {
  __shared__ float sBeta[64 * Eq];   // 32 KB
  __shared__ float sAlpha[64];

  const int m = blockIdx.x;          // 0..2015
  const int p = m % Kq;
  const int bb = m / Kq;
  const int tid = threadIdx.x;
  const int wave = tid >> 5;
  const int lane = tid & 31;

  if (tid == 0) {   // serial softmax over <=63 elements (masked j<=p)
    float mx = -1e30f;
    for (int jj = 0; jj <= p; ++jj) mx = fmaxf(mx, logits[(size_t)m * Kq + jj]);
    float s = 0.0f;
    for (int jj = 0; jj <= p; ++jj) {
      float v = expf(logits[(size_t)m * Kq + jj] - mx);
      sAlpha[jj] = v;
      s += v;
    }
    float inv = 1.0f / s;
    for (int jj = 0; jj < 64; ++jj) sAlpha[jj] = (jj <= p) ? sAlpha[jj] * inv : 0.0f;
  }

  // beta GEMM: A = h_b history [63(+1 pad) x 128] bf16, B = W_bf^T
  const __bf16* A = hb_all + (size_t)m * Kq * Hq;
  for (int t = wave; t < 32; t += 8) {      // 4 row-tiles x 8 col-tiles
    int rt = t >> 3, ct = t & 7;
    floatx8 acc = {};
#pragma unroll
    for (int kk = 0; kk < Hq; kk += 32) {
      int row = rt * 16 + (lane & 15);
      const __bf16* rp = A + (size_t)((row < Kq) ? row : (Kq - 1)) * Hq + kk;
      bf16x16 a = a_frag_bf_guard(rp, row < Kq);
      bf16x16 b = b_frag_wT(Wbf + (size_t)(ct * 16) * Hq + kk, Hq);
      acc = wmma_bf16(a, b, acc);
    }
    store_d_f32(sBeta + (size_t)(rt * 16) * Eq + ct * 16, Eq, acc);
  }
  __syncthreads();

  if (tid < Eq) {
    int e = tid;
    float be = b_bf[e];
    float acc = 0.0f;
    const float* er = emb + (size_t)(bb * Lq) * Eq + e;  // emb[bb*L + i][e]
    for (int jj = 0; jj <= p; ++jj) {
      float bv = tanhf(sBeta[(size_t)jj * Eq + e] + be);
      acc += sAlpha[jj] * bv * er[(size_t)(p - jj) * Eq];
    }
    ctx[(size_t)m * Eq + e] = acc;
    ctx_bf[(size_t)m * Eq + e] = f2bf(acc);
  }
}

// ---------------------------------------------------------------------------
// Kernel 5: preds = ctx @ W_fc^T + b_fc, masked by lengths.
// M=2016 (126 tiles), N=10000 (625 tiles), K=128
// ---------------------------------------------------------------------------
__global__ void final_gemm_kernel(const __bf16* __restrict__ ctx_bf, // [2016,E]
                                  const __bf16* __restrict__ Wfc,    // [V,E] bf16
                                  const float* __restrict__ b_fc,    // [V]
                                  const int* __restrict__ lengths,   // [B]
                                  float* __restrict__ out) {         // [2016,V]
  const int tid  = threadIdx.x;
  const int wave = tid >> 5;
  const int lane = tid & 31;
  const int tile = blockIdx.x * 8 + wave;
  if (tile >= 126 * 625) return;             // wave-uniform
  const int mt = tile / 625, nt = tile % 625;

  const __bf16* arp = ctx_bf + (size_t)(mt * 16 + (lane & 15)) * Eq;
  __builtin_prefetch(Wfc + (size_t)(nt * 16) * Eq, 0, 0);

  floatx8 acc = {};
#pragma unroll
  for (int kk = 0; kk < Eq; kk += 32) {
    bf16x16 a = a_frag_bf(arp + kk);
    bf16x16 b = b_frag_wT(Wfc + (size_t)(nt * 16) * Eq + kk, Eq);
    acc = wmma_bf16(a, b, acc);
  }

  const int col = lane & 15;
  const int r0 = ((lane >> 4) & 1) << 3;
#pragma unroll
  for (int i = 0; i < 8; ++i) {
    int m = mt * 16 + r0 + i;
    int bb = m / Kq, pp = m % Kq;
    bool valid = pp < (lengths[bb] - 1);
    int n = nt * 16 + col;
    out[(size_t)m * Vq + n] = valid ? (acc[i] + b_fc[n]) : 0.0f;
  }
}

// ---------------------------------------------------------------------------
// Host entry
// ---------------------------------------------------------------------------
extern "C" void kernel_launch(void* const* d_in, const int* in_sizes, int n_in,
                              void* d_out, int out_size, void* d_ws, size_t ws_size,
                              hipStream_t stream) {
  (void)in_sizes; (void)n_in; (void)out_size; (void)ws_size;

  // inputs (setup_inputs order); d_in[0] = times is unused by the reference
  const float* codes   = (const float*)d_in[1];
  const int*   lengths = (const int*)  d_in[2];
  const float* W_embed = (const float*)d_in[3];
  const float* W_ih_a  = (const float*)d_in[4];
  const float* W_hh_a  = (const float*)d_in[5];
  const float* b_ih_a  = (const float*)d_in[6];
  const float* b_hh_a  = (const float*)d_in[7];
  const float* W_ih_b  = (const float*)d_in[8];
  const float* W_hh_b  = (const float*)d_in[9];
  const float* b_ih_b  = (const float*)d_in[10];
  const float* b_hh_b  = (const float*)d_in[11];
  const float* W_af    = (const float*)d_in[12];
  const float* b_af    = (const float*)d_in[13];
  const float* W_bf    = (const float*)d_in[14];
  const float* b_bf    = (const float*)d_in[15];
  const float* W_fc    = (const float*)d_in[16];
  const float* b_fc    = (const float*)d_in[17];
  float* out = (float*)d_out;

  // workspace carve-up (~91 MB total), 256B aligned
  char* wsb = (char*)d_ws;
  size_t off = 0;
  auto take = [&](size_t bytes) -> char* {
    char* pp = wsb + off;
    off += (bytes + 255) & ~(size_t)255;
    return pp;
  };
  __bf16* wE       = (__bf16*)take((size_t)Eq * Vq * 2);
  __bf16* wFC      = (__bf16*)take((size_t)Vq * Eq * 2);
  __bf16* wIHa     = (__bf16*)take((size_t)H3q * Eq * 2);
  __bf16* wHHa     = (__bf16*)take((size_t)H3q * Hq * 2);
  __bf16* wIHb     = (__bf16*)take((size_t)H3q * Eq * 2);
  __bf16* wHHb     = (__bf16*)take((size_t)H3q * Hq * 2);
  __bf16* wBF      = (__bf16*)take((size_t)Eq * Hq * 2);
  __bf16* codes_bf = (__bf16*)take((size_t)Bq * Lq * Vq * 2);
  float*  emb      = (float*) take((size_t)Bq * Lq * Eq * 4);
  __bf16* emb_bf   = (__bf16*)take((size_t)Bq * Lq * Eq * 2);
  float*  xpa      = (float*) take((size_t)NBPq * H3q * 4);
  float*  xpb      = (float*) take((size_t)NBPq * H3q * 4);
  float*  ha       = (float*) take((size_t)NBPq * Hq * 4);
  float*  hb       = (float*) take((size_t)NBPq * Hq * 4);
  __bf16* hb_all   = (__bf16*)take((size_t)NBPq * Kq * Hq * 2);
  float*  logits   = (float*) take((size_t)NBPq * Kq * 4);
  float*  ctx      = (float*) take((size_t)NBPq * Eq * 4);
  __bf16* ctxbf    = (__bf16*)take((size_t)NBPq * Eq * 2);

  const dim3 blk(256);
  auto cvt = [&](const float* s, __bf16* d, size_t n) {   // n % 8 == 0
    int n8 = (int)(n / 8);
    convert8_f32_bf16_kernel<<<dim3((n8 + 255) / 256), blk, 0, stream>>>(s, d, n8);
  };

  // 0) inputs/weights -> bf16 (vectorized, memory-bound)
  cvt(codes,   codes_bf, (size_t)Bq * Lq * Vq);
  cvt(W_embed, wE,   (size_t)Eq * Vq);
  cvt(W_fc,    wFC,  (size_t)Vq * Eq);
  cvt(W_ih_a,  wIHa, (size_t)H3q * Eq);
  cvt(W_hh_a,  wHHa, (size_t)H3q * Hq);
  cvt(W_ih_b,  wIHb, (size_t)H3q * Eq);
  cvt(W_hh_b,  wHHb, (size_t)H3q * Hq);
  cvt(W_bf,    wBF,  (size_t)Eq * Hq);

  // 1) embedding GEMM (2048x10000x128)
  embed_gemm_kernel<<<dim3(128), blk, 0, stream>>>(codes_bf, wE, emb, emb_bf);

  // 2) GRU input projections (2016x128x384, both GRUs)
  xproj_gemm_kernel<<<dim3(378, 2), blk, 0, stream>>>(emb_bf, wIHa, wIHb,
                                                      b_ih_a, b_ih_b, xpa, xpb);

  // 3) zero initial hidden states
  zero_f32_kernel<<<dim3((NBPq * Hq / 4 + 255) / 256), blk, 0, stream>>>(ha, NBPq * Hq / 4);
  zero_f32_kernel<<<dim3((NBPq * Hq / 4 + 255) / 256), blk, 0, stream>>>(hb, NBPq * Hq / 4);

  // 4) 63 sequential GRU steps, both GRUs (fused recurrent GEMM + gates)
  for (int j = 0; j < Kq; ++j) {
    gru_step_kernel<<<dim3(NBPq / 32), blk, 0, stream>>>(
        xpa, wHHa, b_ih_a, b_hh_a, ha, W_af, b_af, logits, nullptr, j, 0);
    gru_step_kernel<<<dim3(NBPq / 32), blk, 0, stream>>>(
        xpb, wHHb, b_ih_b, b_hh_b, hb, nullptr, nullptr, nullptr, hb_all, j, 1);
  }

  // 5) softmax + beta GEMM + context (one block per (b,p))
  attn_context_kernel<<<dim3(NBPq), blk, 0, stream>>>(logits, hb_all, wBF,
                                                      b_bf, emb, ctx, ctxbf);

  // 6) output GEMM (2016x10000x128) + length masking
  final_gemm_kernel<<<dim3((126 * 625 + 7) / 8), blk, 0, stream>>>(
      ctxbf, wFC, b_fc, lengths, out);
}
